// MultiHeadAttention_38663295599276
// MI455X (gfx1250) — compile-verified, matrix operand order checked by reading, not run on hardware
//
#include <hip/hip_runtime.h>

// Problem constants (match reference)
#define Bq  4
#define Sq  2048
#define Dq  1024
#define Hq  16
#define DHq 64
#define NEGV (-1e6f)

typedef __attribute__((ext_vector_type(16))) __bf16 v16bf;
typedef __attribute__((ext_vector_type(8)))  __bf16 bf16x8;
typedef __attribute__((ext_vector_type(8)))  float  v8f;
typedef unsigned int u32x4 __attribute__((ext_vector_type(4)));
typedef int          i32x8 __attribute__((ext_vector_type(8)));
typedef int          i32x4 __attribute__((ext_vector_type(4)));

__device__ __forceinline__ v8f wmma_bf16(v16bf a, v16bf b, v8f c) {
  // D(16x16,f32) = A(16x32,bf16) x B(32x16,bf16) + C
  return __builtin_amdgcn_wmma_f32_16x16x32_bf16(false, a, false, b, (short)0, c,
                                                 false, false);
}

// Convert 8 consecutive fp32 -> packed bf16x8 (two b128 global loads)
__device__ __forceinline__ bf16x8 cvt8(const float* __restrict__ p) {
  float4 f0 = *(const float4*)p;
  float4 f1 = *(const float4*)(p + 4);
  bf16x8 r;
  r[0] = (__bf16)f0.x; r[1] = (__bf16)f0.y; r[2] = (__bf16)f0.z; r[3] = (__bf16)f0.w;
  r[4] = (__bf16)f1.x; r[5] = (__bf16)f1.y; r[6] = (__bf16)f1.z; r[7] = (__bf16)f1.w;
  return r;
}

// Load one 16x32 bf16 fragment (A-layout; also usable as B when the tile is
// stored "column rows contiguous", i.e. B[k][n] stored as rows over n).
// Lane L: row = L%16; lanes<16 hold K {0..7,16..23}, lanes>=16 hold K {8..15,24..31}.
__device__ __forceinline__ v16bf load_frag(const __bf16* base, int ld, int lane) {
  const __bf16* p = base + (size_t)(lane & 15) * ld + ((lane >> 4) << 3);
  bf16x8 lo = *(const bf16x8*)p;        // K base .. base+7
  bf16x8 hi = *(const bf16x8*)(p + 16); // K base+16 .. base+23
  v16bf r;
#pragma unroll
  for (int i = 0; i < 8; ++i) { r[i] = lo[i]; r[8 + i] = hi[i]; }
  return r;
}

// ---------------------------------------------------------------------------
// TDM: async-load a 64x64 bf16 tile (row stride Dq elems) from global -> LDS.
// 2-D descriptor per CDNA5 ISA:
//   group0: count=1 | lds_addr | global_addr | type=2
//   group1: data_size=1(2B), tensor_dim0=64, tensor_dim1=Sq,
//           tile_dim0=64, tile_dim1=64, tensor_dim0_stride=Dq
// Issued once per workgroup (wave 0); tracked with TENSORcnt.
// ---------------------------------------------------------------------------
__device__ __forceinline__ void tdm_load_tile64x64(const __bf16* gbase, void* lds_dst) {
  const unsigned long long ga = (unsigned long long)(uintptr_t)gbase;
  const unsigned int       la = (unsigned int)(uintptr_t)lds_dst;
  u32x4 g0;
  g0[0] = 1u;                                              // count=1, user mode
  g0[1] = la;                                              // lds_addr (bytes)
  g0[2] = (unsigned int)ga;                                // global_addr[31:0]
  g0[3] = (unsigned int)((ga >> 32) & 0x01FFFFFFu) | 0x80000000u; // [56:32] | type=2
  i32x8 g1;
  g1[0] = 0x00010000;            // workgroup_mask=0, data_size=1 (2 bytes)
  g1[1] = (int)(64u << 16);      // tensor_dim0 = 64 (bits 79:48, low half)
  g1[2] = (int)((unsigned)Sq << 16); // tensor_dim1 = 2048 (bits 111:80, low half)
  g1[3] = (int)(64u << 16);      // tile_dim0 = 64 (bits 127:112)
  g1[4] = 64;                    // tile_dim1 = 64 (bits 143:128); tile_dim2 = 0
  g1[5] = Dq;                    // tensor_dim0_stride = 1024 (bits 207:160, low 32)
  g1[6] = 0;                     // stride0 hi / stride1 lo
  g1[7] = 0;
  i32x4 z4 = {0, 0, 0, 0};       // groups 2/3 unused (2-D tile: tile_dim2/3 = 0)
#if __clang_major__ >= 23
  i32x8 z8 = {0, 0, 0, 0, 0, 0, 0, 0};
  __builtin_amdgcn_tensor_load_to_lds(g0, g1, z4, z4, z8, 0);
#else
  __builtin_amdgcn_tensor_load_to_lds(g0, g1, z4, z4, 0);
#endif
}

// ---------------------------------------------------------------------------
// C[M,N] = A[M,K] @ W[N,K]^T   (torch Linear: y = x @ W.T)
// Block tile 128x128, 8 waves (256 thr), each wave 32x64. K staged 64 wide.
// TA = float (convert to bf16 while staging) or __bf16 (direct copy).
// TC = __bf16 (intermediate) or float (final output).
// ---------------------------------------------------------------------------
template <typename TA, typename TC>
__global__ __launch_bounds__(256) void gemm_xwT(const TA* __restrict__ A,
                                                const float* __restrict__ W,
                                                TC* __restrict__ C,
                                                int M, int N, int K) {
  __shared__ __align__(16) __bf16 sA[128 * 64];
  __shared__ __align__(16) __bf16 sB[128 * 64];

  const int tid  = threadIdx.x;
  const int lane = tid & 31;
  const int wave = tid >> 5;       // 0..7
  const int wm   = wave & 3;       // rows += 32*wm
  const int wn   = wave >> 2;      // cols += 64*wn
  const int bm   = (int)blockIdx.x * 128;
  const int bn   = (int)blockIdx.y * 128;

  v8f acc[2][4];
  const v8f vzero = {0.f, 0.f, 0.f, 0.f, 0.f, 0.f, 0.f, 0.f};
#pragma unroll
  for (int i = 0; i < 2; ++i)
#pragma unroll
    for (int j = 0; j < 4; ++j) acc[i][j] = vzero;

  const int srow = tid >> 1;            // 0..127
  const int sseg = (tid & 1) * 32;      // 0 / 32

  for (int k0 = 0; k0 < K; k0 += 64) {
    // Stage A tile (128x64) and W tile (128x64) as bf16
    const TA*    ap = A + (size_t)(bm + srow) * K + k0 + sseg;
    const float* wp = W + (size_t)(bn + srow) * K + k0 + sseg;
#pragma unroll
    for (int j = 0; j < 4; ++j) {
      bf16x8 pa;
      if constexpr (__is_same(TA, float)) {
        pa = cvt8((const float*)ap + j * 8);
      } else {
        pa = *(const bf16x8*)(ap + j * 8);
      }
      *(bf16x8*)&sA[srow * 64 + sseg + j * 8] = pa;
      *(bf16x8*)&sB[srow * 64 + sseg + j * 8] = cvt8(wp + j * 8);
    }
    if (k0 + 64 < K) {  // pull next tiles toward GL2 while we compute
      __builtin_prefetch(ap + 64, 0, 1);
      __builtin_prefetch(wp + 64, 0, 1);
    }
    __syncthreads();

    v16bf af[2][2];
#pragma unroll
    for (int i = 0; i < 2; ++i)
#pragma unroll
      for (int kc = 0; kc < 2; ++kc)
        af[i][kc] = load_frag(sA + (wm * 32 + i * 16) * 64 + kc * 32, 64, lane);

#pragma unroll
    for (int kc = 0; kc < 2; ++kc)
#pragma unroll
      for (int j = 0; j < 4; ++j) {
        v16bf bfrag = load_frag(sB + (wn * 64 + j * 16) * 64 + kc * 32, 64, lane);
#pragma unroll
        for (int i = 0; i < 2; ++i) acc[i][j] = wmma_bf16(af[i][kc], bfrag, acc[i][j]);
      }

    __syncthreads();
  }

  // Epilogue: C layout — VGPR r: lanes 0-15 -> M=r, lanes 16-31 -> M=r+8
  const int hi8 = (lane >> 4) * 8;
  const int col = lane & 15;
#pragma unroll
  for (int i = 0; i < 2; ++i)
#pragma unroll
    for (int j = 0; j < 4; ++j)
#pragma unroll
      for (int r = 0; r < 8; ++r) {
        int m = bm + wm * 32 + i * 16 + r + hi8;
        int n = bn + wn * 64 + j * 16 + col;
        C[(size_t)m * N + n] = (TC)acc[i][j][r];
      }
}

// ---------------------------------------------------------------------------
// Flash-style attention. Grid: (S/64, H, B); 128 threads = 4 waves,
// each wave owns 16 q-rows. Streams 64-wide KV blocks, online softmax.
// K tile staged by the Tensor Data Mover (async, TENSORcnt); V tile staged
// transposed by the threads while the TDM runs.
// ---------------------------------------------------------------------------
__global__ __launch_bounds__(128) void attn_kernel(const __bf16* __restrict__ Q,
                                                   const __bf16* __restrict__ Kp,
                                                   const __bf16* __restrict__ Vp,
                                                   const int* __restrict__ valid_lens,
                                                   __bf16* __restrict__ O) {
  __shared__ __align__(16) __bf16 sK[64 * 64];   // [kv][d]   (TDM destination)
  __shared__ __align__(16) __bf16 sVt[64 * 64];  // [d][kv]   (transposed)
  __shared__ __align__(16) __bf16 sP[4 * 16 * 64];

  const int tid  = threadIdx.x;
  const int lane = tid & 31;
  const int wave = tid >> 5;
  const int b    = blockIdx.z;
  const int h    = blockIdx.y;
  const int q0   = (int)blockIdx.x * 64 + wave * 16;
  const int vl   = valid_lens[b];

  const size_t headoff = (size_t)b * Sq * Dq + h * DHq;

  // Q fragments live in registers for the whole kernel (16 rows x 64 dh)
  const __bf16* qbase = Q + headoff + (size_t)q0 * Dq;
  v16bf aq0 = load_frag(qbase, Dq, lane);
  v16bf aq1 = load_frag(qbase + 32, Dq, lane);

  v8f acc[4];
  const v8f vzero = {0.f, 0.f, 0.f, 0.f, 0.f, 0.f, 0.f, 0.f};
#pragma unroll
  for (int d = 0; d < 4; ++d) acc[d] = vzero;
  float m_r[8], l_r[8];
#pragma unroll
  for (int r = 0; r < 8; ++r) { m_r[r] = -1e30f; l_r[r] = 0.f; }

  const int hi8 = (lane >> 4) * 8;
  const int col = lane & 15;
  __bf16* sPw = sP + wave * 16 * 64;

  // Blocks fully past valid_len contribute exactly 0 (exp underflow) -> skip.
  const int kv_end = ((vl + 63) & ~63) < Sq ? ((vl + 63) & ~63) : Sq;

  for (int kv0 = 0; kv0 < kv_end; kv0 += 64) {
    // --- K tile via TDM (one async DMA per workgroup, overlaps V staging) --
    if (wave == 0)
      tdm_load_tile64x64(Kp + headoff + (size_t)kv0 * Dq, &sK[0]);

    // --- V tile staged transposed by all threads --------------------------
    {
      const int krow  = tid >> 1;        // 0..63
      const int khalf = (tid & 1) * 32;  // 0 / 32
      const __bf16* vp = Vp + headoff + (size_t)(kv0 + krow) * Dq + khalf;
#pragma unroll
      for (int j = 0; j < 4; ++j) {
        bf16x8 v = *(const bf16x8*)(vp + j * 8);
#pragma unroll
        for (int e = 0; e < 8; ++e) sVt[(khalf + j * 8 + e) * 64 + krow] = v[e];
      }
    }
    if (wave == 0) __builtin_amdgcn_s_wait_tensorcnt((short)0);
    __syncthreads();

    // --- scores: S(16x64) = Q(16x64) . K^T, scaled + masked ------------
    v8f sc[4];
#pragma unroll
    for (int n = 0; n < 4; ++n) {
      v16bf bk0 = load_frag(sK + (n * 16) * 64, 64, lane);
      v16bf bk1 = load_frag(sK + (n * 16) * 64 + 32, 64, lane);
      v8f s = vzero;
      s = wmma_bf16(aq0, bk0, s);
      s = wmma_bf16(aq1, bk1, s);
      const bool valid = (kv0 + n * 16 + col) < vl;
#pragma unroll
      for (int r = 0; r < 8; ++r) sc[n][r] = valid ? s[r] * 0.125f : NEGV;
    }

    // --- online softmax (rows split per half-wave: lanes<16 rows r, >=16 r+8)
    float mnew[8], alpha[8];
#pragma unroll
    for (int r = 0; r < 8; ++r) {
      float x = fmaxf(fmaxf(sc[0][r], sc[1][r]), fmaxf(sc[2][r], sc[3][r]));
      x = fmaxf(x, __shfl_xor(x, 1));
      x = fmaxf(x, __shfl_xor(x, 2));
      x = fmaxf(x, __shfl_xor(x, 4));
      x = fmaxf(x, __shfl_xor(x, 8));
      mnew[r]  = fmaxf(m_r[r], x);
      alpha[r] = __expf(m_r[r] - mnew[r]);
      m_r[r]   = mnew[r];
    }
#pragma unroll
    for (int n = 0; n < 4; ++n)
#pragma unroll
      for (int r = 0; r < 8; ++r) sc[n][r] = __expf(sc[n][r] - mnew[r]);
#pragma unroll
    for (int r = 0; r < 8; ++r) {
      float s = sc[0][r] + sc[1][r] + sc[2][r] + sc[3][r];
      s += __shfl_xor(s, 1);
      s += __shfl_xor(s, 2);
      s += __shfl_xor(s, 4);
      s += __shfl_xor(s, 8);
      l_r[r] = l_r[r] * alpha[r] + s;
    }
#pragma unroll
    for (int d = 0; d < 4; ++d)
#pragma unroll
      for (int r = 0; r < 8; ++r) acc[d][r] *= alpha[r];

    // --- P: C-layout -> A-layout via per-wave LDS ----------------------
#pragma unroll
    for (int n = 0; n < 4; ++n)
#pragma unroll
      for (int r = 0; r < 8; ++r)
        sPw[(r + hi8) * 64 + n * 16 + col] = (__bf16)sc[n][r];
    __syncthreads();

    // --- O += P(16x64) . V(64x64) --------------------------------------
    v16bf pa0 = load_frag(sPw, 64, lane);
    v16bf pa1 = load_frag(sPw + 32, 64, lane);
#pragma unroll
    for (int d = 0; d < 4; ++d) {
      v16bf bv0 = load_frag(sVt + (d * 16) * 64, 64, lane);
      v16bf bv1 = load_frag(sVt + (d * 16) * 64 + 32, 64, lane);
      acc[d] = wmma_bf16(pa0, bv0, acc[d]);
      acc[d] = wmma_bf16(pa1, bv1, acc[d]);
    }
    __syncthreads();  // before next block overwrites sK/sVt
  }

  // --- normalize and store (bf16, fed to output projection) ------------
  float inv[8];
#pragma unroll
  for (int r = 0; r < 8; ++r) inv[r] = 1.0f / l_r[r];
#pragma unroll
  for (int d = 0; d < 4; ++d)
#pragma unroll
    for (int r = 0; r < 8; ++r) {
      int row = q0 + r + hi8;
      O[(size_t)(b * Sq + row) * Dq + h * DHq + d * 16 + col] =
          (__bf16)(acc[d][r] * inv[r]);
    }
}

// ---------------------------------------------------------------------------
extern "C" void kernel_launch(void* const* d_in, const int* in_sizes, int n_in,
                              void* d_out, int out_size, void* d_ws, size_t ws_size,
                              hipStream_t stream) {
  const float* queries    = (const float*)d_in[0];
  const float* keys       = (const float*)d_in[1];
  const float* values     = (const float*)d_in[2];
  const int*   valid_lens = (const int*)d_in[3];
  const float* W_q        = (const float*)d_in[4];
  const float* W_k        = (const float*)d_in[5];
  const float* W_v        = (const float*)d_in[6];
  const float* W_o        = (const float*)d_in[7];
  float* out = (float*)d_out;

  const size_t NE = (size_t)Bq * Sq * Dq;  // 8M elems
  __bf16* Qp = (__bf16*)d_ws;
  __bf16* Kp = Qp + NE;
  __bf16* Vp = Kp + NE;
  __bf16* AO = Vp + NE;

  const int M = Bq * Sq;  // 8192
  dim3 gg(M / 128, Dq / 128);

  gemm_xwT<float, __bf16><<<gg, 256, 0, stream>>>(queries, W_q, Qp, M, Dq, Dq);
  gemm_xwT<float, __bf16><<<gg, 256, 0, stream>>>(keys,    W_k, Kp, M, Dq, Dq);
  gemm_xwT<float, __bf16><<<gg, 256, 0, stream>>>(values,  W_v, Vp, M, Dq, Dq);

  attn_kernel<<<dim3(Sq / 64, Hq, Bq), 128, 0, stream>>>(Qp, Kp, Vp, valid_lens, AO);

  gemm_xwT<__bf16, float><<<gg, 256, 0, stream>>>(AO, W_o, out, M, Dq, Dq);
}
